// MoE_72670846648535
// MI455X (gfx1250) — matrix operand order repaired and checked by reading, not compiled, not run on hardware
//
#include <hip/hip_runtime.h>

// MoE forward: out[B,E,C] + scores[B,E];  B=32768, IN=HID=C=128, E=32.

#define B_TOK 32768
#define IN_D  128
#define HID_D 128
#define NE    32
#define NC    128
#define ASTR  136   // LDS row stride in halves: 272B = 16B-aligned rows, bank-conflict free

typedef __attribute__((ext_vector_type(16))) _Float16 v16h;
typedef __attribute__((ext_vector_type(8)))  float    v8f;

union AF { v16h h; unsigned int u[8]; };

__device__ __forceinline__ unsigned short f2h_bits(float f) {
  _Float16 hv = (_Float16)f;
  return __builtin_bit_cast(unsigned short, hv);
}

// ---------------- Router: scores = softmax(relu(x @ Wg + bg)) ----------------
__launch_bounds__(256)
__global__ void moe_router(const float* __restrict__ x, const float* __restrict__ Wg,
                           const float* __restrict__ bg, float* __restrict__ scores) {
  const int lane = threadIdx.x & 31;
  const int tok  = blockIdx.x * 8 + (threadIdx.x >> 5);
  const float* xr = x + (size_t)tok * IN_D;
  float v = bg[lane];
#pragma unroll 8
  for (int i = 0; i < IN_D; ++i) v = fmaf(xr[i], Wg[i * NE + lane], v);
  v = fmaxf(v, 0.f);
  float m = v;
#pragma unroll
  for (int off = 16; off > 0; off >>= 1) m = fmaxf(m, __shfl_xor(m, off, 32));
  float ev = __expf(v - m);
  float s = ev;
#pragma unroll
  for (int off = 16; off > 0; off >>= 1) s += __shfl_xor(s, off, 32);
  scores[(size_t)tok * NE + lane] = ev / s;
}

// ---------------- Prep: f32 -> f16 staging in workspace ----------------
__launch_bounds__(256)
__global__ void prep_x(const float* __restrict__ x, unsigned short* __restrict__ xh) {
  size_t i = (size_t)blockIdx.x * 256 + threadIdx.x;   // B*128 threads
  xh[i] = f2h_bits(x[i]);
}

// Transpose each 128x128 matrix: wh[m][n*128+k] = f16(W[m][k*128+n])
__launch_bounds__(256)
__global__ void prep_w(const float* __restrict__ W, unsigned short* __restrict__ wh) {
  int m   = blockIdx.x >> 6;                  // 64 blocks x 256 threads per matrix
  int idx = ((blockIdx.x & 63) << 8) + threadIdx.x;
  int k = idx >> 7, n = idx & 127;
  wh[(size_t)m * 16384 + n * 128 + k] = f2h_bits(W[(size_t)m * 16384 + idx]);
}

// ---------------- async global->LDS copy (CDNA5 ASYNCcnt path) ----------------
// Copies 128 rows of 128 halves (256B) from global (row stride 128 halves) into
// LDS (row stride ASTR halves). Per wave: 16 rows => 8 b128 instructions
// (each instruction: 32 lanes x 16B = 2 rows).
__device__ __forceinline__ void async_copy_tile(unsigned ldsBase,
                                                const unsigned short* __restrict__ src,
                                                int wv, int lane) {
  int r0 = wv * 16 + (lane >> 4);       // this lane's row (2 rows / instruction)
  int c8 = lane & 15;                   // 16-byte chunk within row
  const unsigned short* g = src + (size_t)r0 * 128 + c8 * 8;
  unsigned l = ldsBase + (unsigned)(r0 * ASTR + c8 * 8) * 2;
#pragma unroll
  for (int j = 0; j < 8; ++j) {
    asm volatile("global_load_async_to_lds_b128 %0, %1, off"
                 :: "v"(l), "v"(g) : "memory");
    g += 2 * 128;
    l += 2 * ASTR * 2;
  }
}

#define WAIT_ASYNC(n) asm volatile("s_wait_asynccnt " #n ::: "memory")

// ---------------- shared GEMM phase ----------------
// D[128x128] = act @ W + bias (optional ReLU); act: LDS [token][k] f16,
// wT: LDS [n][k] f16, both stride ASTR. Output to LDS f16 or global f32.
template<bool RELU, bool TO_LDS>
__device__ __forceinline__ void gemm_phase(const unsigned short* __restrict__ act,
                                           const unsigned short* __restrict__ wT,
                                           const float* __restrict__ bias,
                                           unsigned short* __restrict__ dstL,
                                           float* __restrict__ dstG, int gstride) {
  const int lane = threadIdx.x & 31;
  const int wv   = threadIdx.x >> 5;
  const int mr   = lane & 15;
  const int g    = lane >> 4;

  AF a[4];   // A fragments per ISA 16-bit A layout
  const unsigned short* arow = act + (wv * 16 + mr) * ASTR;
#pragma unroll
  for (int kk = 0; kk < 4; ++kk) {
#pragma unroll
    for (int v = 0; v < 8; ++v) {
      int k = kk * 32 + ((v < 4) ? 0 : 16) + g * 8 + (v & 3) * 2;
      a[kk].u[v] = *(const unsigned int*)(arow + k);
    }
  }

#pragma unroll
  for (int nt = 0; nt < 8; ++nt) {
    float bval = bias[nt * 16 + mr];
    v8f acc;
#pragma unroll
    for (int r = 0; r < 8; ++r) acc[r] = bval;

    const unsigned short* wrow = wT + (nt * 16 + mr) * ASTR;
#pragma unroll
    for (int kk = 0; kk < 4; ++kk) {
      AF bm;  // B fragment: lane=N, half-wave selects K range
#pragma unroll
      for (int v = 0; v < 8; ++v)
        bm.u[v] = *(const unsigned int*)(wrow + kk * 32 + ((v < 4) ? 0 : 16) + g * 8 + (v & 3) * 2);
      acc = __builtin_amdgcn_wmma_f32_16x16x32_f16(
          false, a[kk].h, false, bm.h, (short)0, acc, false, false);
    }
    if (RELU) {
#pragma unroll
      for (int r = 0; r < 8; ++r) acc[r] = fmaxf(acc[r], 0.f);
    }
    if (TO_LDS) {
#pragma unroll
      for (int r = 0; r < 8; ++r) {
        int row = wv * 16 + r + 8 * g;
        dstL[row * ASTR + nt * 16 + mr] = f2h_bits(acc[r]);
      }
    } else {
#pragma unroll
      for (int r = 0; r < 8; ++r) {
        int row = wv * 16 + r + 8 * g;
        dstG[(size_t)row * gstride + nt * 16 + mr] = acc[r];
      }
    }
  }
}

// ---------------- Main kernel: async-staged, double-buffered weights ----------------
__launch_bounds__(256)
__global__ void moe_experts_async(const unsigned short* __restrict__ xh,
                                  const unsigned short* __restrict__ w1h,
                                  const unsigned short* __restrict__ w2h,
                                  const unsigned short* __restrict__ wch,
                                  const float* __restrict__ b1, const float* __restrict__ b2,
                                  const float* __restrict__ bc, float* __restrict__ out) {
  __shared__ __align__(16) unsigned short sW0[128 * ASTR];
  __shared__ __align__(16) unsigned short sW1[128 * ASTR];
  __shared__ __align__(16) unsigned short sA [128 * ASTR];
  __shared__ __align__(16) unsigned short sB [128 * ASTR];

  const int e    = blockIdx.y;
  const int tok0 = blockIdx.x * 128;
  const int lane = threadIdx.x & 31;
  const int wv   = threadIdx.x >> 5;

  const unsigned lA  = (unsigned)(uintptr_t)(void*)sA;
  const unsigned lW0 = (unsigned)(uintptr_t)(void*)sW0;
  const unsigned lW1 = (unsigned)(uintptr_t)(void*)sW1;

  // Kick everything: X, W1, then W2 (in-order completion per wave).
  async_copy_tile(lA,  xh  + (size_t)tok0 * 128, wv, lane);   // 8 instrs
  async_copy_tile(lW0, w1h + (size_t)e * 16384,  wv, lane);   // 8 instrs
  async_copy_tile(lW1, w2h + (size_t)e * 16384,  wv, lane);   // 8 instrs

  WAIT_ASYNC(8);        // X + W1 landed (W2 may still be in flight)
  __syncthreads();

  // Phase 1: H = relu(X @ W1 + b1)   (sA,sW0 -> sB); W2 streams in behind it.
  gemm_phase<true, true>(sA, sW0, b1 + e * 128, sB, nullptr, 0);
  __syncthreads();      // everyone done reading sW0 / writing sB

  // Prefetch Wc into the W1 buffer while phase 2 runs.
  async_copy_tile(lW0, wch, wv, lane);
  WAIT_ASYNC(8);        // W2 fully landed (only Wc's 8 may remain)
  __syncthreads();

  // Phase 2: EO = H @ W2 + b2        (sB,sW1 -> sA)
  gemm_phase<false, true>(sB, sW1, b2 + e * 128, sA, nullptr, 0);
  __syncthreads();

  WAIT_ASYNC(0);        // Wc landed
  __syncthreads();

  // Phase 3: OUT = EO @ Wc + bc      (sA,sW0 -> global)
  float* gout = out + ((size_t)tok0 * NE + e) * NC;
  gemm_phase<false, false>(sA, sW0, bc, nullptr, gout, NE * NC);
}

// ---------------- Fallback: synchronous staging (no workspace needed) ----------------
__device__ __forceinline__ void load_w_t(unsigned short* sW, const float* __restrict__ W, int t) {
#pragma unroll
  for (int j = 0; j < 64; ++j) {
    int idx = t + j * 256;
    int k = idx >> 7, n = idx & 127;
    sW[n * ASTR + k] = f2h_bits(W[idx]);
  }
}

__launch_bounds__(256)
__global__ void moe_experts_sync(const float* __restrict__ x,
                                 const float* __restrict__ W1, const float* __restrict__ b1,
                                 const float* __restrict__ W2, const float* __restrict__ b2,
                                 const float* __restrict__ Wc, const float* __restrict__ bc,
                                 float* __restrict__ out) {
  __shared__ __align__(16) unsigned short sW[128 * ASTR];
  __shared__ __align__(16) unsigned short sA[128 * ASTR];
  __shared__ __align__(16) unsigned short sB[128 * ASTR];

  const int e    = blockIdx.y;
  const int tok0 = blockIdx.x * 128;
  const int t    = threadIdx.x;

  const float* xsrc = x + (size_t)tok0 * IN_D;
#pragma unroll
  for (int j = 0; j < 64; ++j) {
    int idx = t + j * 256;
    sA[(idx >> 7) * ASTR + (idx & 127)] = f2h_bits(xsrc[idx]);
  }
  load_w_t(sW, W1 + (size_t)e * 16384, t);
  __syncthreads();
  gemm_phase<true, true>(sA, sW, b1 + e * 128, sB, nullptr, 0);
  __syncthreads();
  load_w_t(sW, W2 + (size_t)e * 16384, t);
  __syncthreads();
  gemm_phase<false, true>(sB, sW, b2 + e * 128, sA, nullptr, 0);
  __syncthreads();
  load_w_t(sW, Wc, t);
  __syncthreads();
  float* gout = out + ((size_t)tok0 * NE + e) * NC;
  gemm_phase<false, false>(sA, sW, bc, nullptr, gout, NE * NC);
}

// ---------------- host launch ----------------
extern "C" void kernel_launch(void* const* d_in, const int* in_sizes, int n_in,
                              void* d_out, int out_size, void* d_ws, size_t ws_size,
                              hipStream_t stream) {
  (void)in_sizes; (void)n_in; (void)out_size;
  const float* x  = (const float*)d_in[0];
  const float* Wg = (const float*)d_in[1];
  const float* bg = (const float*)d_in[2];
  const float* W1 = (const float*)d_in[3];
  const float* b1 = (const float*)d_in[4];
  const float* W2 = (const float*)d_in[5];
  const float* b2 = (const float*)d_in[6];
  const float* Wc = (const float*)d_in[7];
  const float* bc = (const float*)d_in[8];

  float* out    = (float*)d_out;
  float* scores = out + (size_t)B_TOK * NE * NC;

  moe_router<<<B_TOK / 8, 256, 0, stream>>>(x, Wg, bg, scores);

  // workspace layout (halves): xh[B*128] | w1h[32*16384] | w2h[32*16384] | wch[16384]
  const size_t nXh = (size_t)B_TOK * 128;
  const size_t nW  = (size_t)NE * 16384;
  const size_t REQ = (nXh + 2 * nW + 16384) * sizeof(unsigned short);

  dim3 grid(B_TOK / 128, NE);
  if (ws_size >= REQ) {
    unsigned short* xh  = (unsigned short*)d_ws;
    unsigned short* w1h = xh + nXh;
    unsigned short* w2h = w1h + nW;
    unsigned short* wch = w2h + nW;
    prep_x<<<(B_TOK * 128) / 256, 256, 0, stream>>>(x, xh);
    prep_w<<<NE * 64, 256, 0, stream>>>(W1, w1h);
    prep_w<<<NE * 64, 256, 0, stream>>>(W2, w2h);
    prep_w<<<64, 256, 0, stream>>>(Wc, wch);
    moe_experts_async<<<grid, 256, 0, stream>>>(xh, w1h, w2h, wch, b1, b2, bc, out);
  } else {
    moe_experts_sync<<<grid, 256, 0, stream>>>(x, W1, b1, W2, b2, Wc, bc, out);
  }
}
